// NodeModel_47966194762017
// MI455X (gfx1250) — matrix operand order repaired and checked by reading, not compiled, not run on hardware
//
#include <hip/hip_runtime.h>

typedef __attribute__((ext_vector_type(16))) _Float16 v16h;
typedef __attribute__((ext_vector_type(8)))  _Float16 v8h_t;
typedef __attribute__((ext_vector_type(8)))  float    v8f;

#define WMMA_F16(A, B, C) \
  __builtin_amdgcn_wmma_f32_16x16x32_f16(false, (A), false, (B), (short)0, (C), false, false)

__device__ __forceinline__ void atomAddF(float* p, float v) {
  unsafeAtomicAdd(p, v);   // lowers to global_atomic_add_f32 on gfx1250
}

// ---------------------------------------------------------------------------
// Zero the scatter sums + counts (required every launch: harness poisons ws).
// ---------------------------------------------------------------------------
__global__ void zero_kernel(float* __restrict__ p, long n) {
  long i = blockIdx.x * (long)blockDim.x + threadIdx.x;
  long stride = (long)gridDim.x * blockDim.x;
  for (; i < n; i += stride) p[i] = 0.0f;
}

// ---------------------------------------------------------------------------
// Pre-pack an f32 weight [fin, fout] into WMMA B-fragment-ordered f16 buffer.
// Fragment id = t*ksteps + s (t = N-tile of 16, s = K-step of 32).
// Within a fragment: 32 lanes x 16 halves; lane group g = lane/16 holds
// K = s*32 + 8*g + (e<8 ? e : e+8), column n = t*16 + lane%16.  K >= fin is zero padding.
// ---------------------------------------------------------------------------
__global__ void pack_kernel(const float* __restrict__ src, _Float16* __restrict__ dst,
                            int fin, int fout, int ksteps, int total) {
  int i = blockIdx.x * blockDim.x + threadIdx.x;
  if (i >= total) return;
  int frag   = i >> 9;          // /512 halves per fragment
  int within = i & 511;
  int lane   = within >> 4;
  int e      = within & 15;
  int s      = frag % ksteps;
  int t      = frag / ksteps;
  int K      = s * 32 + 8 * (lane >> 4) + (e < 8 ? e : e + 8);
  int n      = t * 16 + (lane & 15);
  float v    = (K < fin) ? src[(size_t)K * fout + n] : 0.0f;
  dst[i] = (_Float16)v;
}

// ---------------------------------------------------------------------------
// Edge kernel: per wave, a 16-edge tile.
//   e_in[16,4] -> (WMMA, K padded to 32) -> relu -> [16,64]
//   -> LDS restage -> (WMMA K=64) -> [16,128] + b1b -> atomic scatter-add.
// ---------------------------------------------------------------------------
__global__ __launch_bounds__(128) void edge_kernel(
    const float* __restrict__ x, const int* __restrict__ eidx,
    const float* __restrict__ eattr,
    const float* __restrict__ b1a, const float* __restrict__ b1b,
    const _Float16* __restrict__ fW1a, const _Float16* __restrict__ fW1b,
    float* __restrict__ sums, float* __restrict__ cnt, int E) {
  __shared__ _Float16 h1[4][16][64];
  __shared__ int      rix[4][16];

  const int wave = threadIdx.x >> 5;
  const int lane = threadIdx.x & 31;
  const int tile = blockIdx.x * 4 + wave;
  const long base = (long)tile * 16;
  if (base >= E) return;                 // wave-uniform
  const int g = lane >> 4;
  const int n = lane & 15;

  // A fragment: row M = lane (lanes 0-15 hold K=0..7,16..23 -> only K<4 used).
  v16h a1 = {};
  if (lane < 16) {
    long e = base + lane;
    if (e < E) {
      int col = eidx[E + e];             // edge_index[1][e]
      int row = eidx[e];                 // edge_index[0][e]
      a1[0] = (_Float16)x[(long)col * 2 + 0];
      a1[1] = (_Float16)x[(long)col * 2 + 1];
      a1[2] = (_Float16)eattr[e * 2 + 0];
      a1[3] = (_Float16)eattr[e * 2 + 1];
      rix[wave][lane] = row;
      atomAddF(&cnt[row], 1.0f);
    } else {
      rix[wave][lane] = -1;
    }
  }

  // GEMM1: [16,32] x [32,64] -> relu -> stage f16 into LDS (A-layout source).
#pragma unroll
  for (int t = 0; t < 4; ++t) {
    v8f c;
    float bb = b1a[t * 16 + n];
#pragma unroll
    for (int j = 0; j < 8; ++j) c[j] = bb;
    v16h b = ((const v16h*)(fW1a + (size_t)t * 512))[lane];
    c = WMMA_F16(a1, b, c);
#pragma unroll
    for (int j = 0; j < 8; ++j) {
      int M = j + 8 * g;                 // C/D layout: M = vgpr + 8*(lane/16)
      float v = c[j];
      h1[wave][M][t * 16 + n] = (_Float16)(v > 0.0f ? v : 0.0f);
    }
  }

  // Reload as A fragments (K=64 -> 2 k-steps). A layout: lane row M = lane%16,
  // element e -> K = kstep*32 + 8*g + (e<8 ? e : e+8)  => two 16B LDS loads.
  v16h a2[2];
#pragma unroll
  for (int s = 0; s < 2; ++s) {
    const _Float16* rp = &h1[wave][n][0];
    int K0 = s * 32 + 8 * g;
    v8h_t lo = *(const v8h_t*)(rp + K0);
    v8h_t hi = *(const v8h_t*)(rp + K0 + 16);
#pragma unroll
    for (int i = 0; i < 8; ++i) { a2[s][i] = lo[i]; a2[s][i + 8] = hi[i]; }
  }

  // GEMM2: [16,64] x [64,128] + b1b, then scatter-add rows into sums.
#pragma unroll
  for (int t = 0; t < 8; ++t) {
    v8f d;
    float bb = b1b[t * 16 + n];
#pragma unroll
    for (int j = 0; j < 8; ++j) d[j] = bb;
#pragma unroll
    for (int s = 0; s < 2; ++s) {
      v16h b = ((const v16h*)(fW1b + (size_t)(t * 2 + s) * 512))[lane];
      d = WMMA_F16(a2[s], b, d);
    }
#pragma unroll
    for (int j = 0; j < 8; ++j) {
      int M = j + 8 * g;
      int r = rix[wave][M];
      if (r >= 0) atomAddF(&sums[(long)r * 128 + t * 16 + n], d[j]);
    }
  }
}

// ---------------------------------------------------------------------------
// Node kernel: per wave, a 16-node tile.
//   z[16,132] (pad K->160) -> (5 WMMA ksteps x 16 ntiles) -> relu ->
//   restage -> (8 ksteps x 32 ntiles) -> out[16,512] f32.
// ---------------------------------------------------------------------------
__global__ __launch_bounds__(128) void node_kernel(
    const float* __restrict__ x, const float* __restrict__ u,
    const int* __restrict__ batch,
    const float* __restrict__ sums, const float* __restrict__ cnt,
    const float* __restrict__ b2a, const float* __restrict__ b2b,
    const _Float16* __restrict__ fW2a, const _Float16* __restrict__ fW2b,
    float* __restrict__ out, int N) {
  __shared__ _Float16 zh[4][16][256];    // z (cols 0..159) then h1 (cols 0..255)

  const int wave = threadIdx.x >> 5;
  const int lane = threadIdx.x & 31;
  const int tile = blockIdx.x * 4 + wave;
  if ((long)tile * 16 >= N) return;      // wave-uniform
  const int g = lane >> 4;
  const int n = lane & 15;

  // Build z row M = lane%16, column half g*80..g*80+79 (f16, zero-padded).
  {
    long nd = (long)tile * 16 + n;
    bool valid = nd < N;
    float inv = 0.0f;
    long bidx = 0;
    if (valid) {
      inv = 1.0f / fmaxf(cnt[nd], 1.0f);
      bidx = batch[nd];
    }
    int c0 = g * 80;
    for (int c = c0; c < c0 + 80; ++c) {
      float v = 0.0f;
      if (valid) {
        if (c < 2)        v = x[nd * 2 + c];
        else if (c < 130) v = sums[nd * 128 + (c - 2)] * inv;
        else if (c < 132) v = u[bidx * 2 + (c - 130)];
      }
      zh[wave][n][c] = (_Float16)v;
    }
  }

  // Preload 5 A fragments of z (row stride 256 halves).
  v16h az[5];
#pragma unroll
  for (int s = 0; s < 5; ++s) {
    const _Float16* rp = &zh[wave][n][0];
    int K0 = s * 32 + 8 * g;
    v8h_t lo = *(const v8h_t*)(rp + K0);
    v8h_t hi = *(const v8h_t*)(rp + K0 + 16);
#pragma unroll
    for (int i = 0; i < 8; ++i) { az[s][i] = lo[i]; az[s][i + 8] = hi[i]; }
  }

  // GEMM1: 16 N-tiles x 5 K-steps, bias-initialized accumulators.
  v8f c1[16];
#pragma unroll
  for (int t = 0; t < 16; ++t) {
    float bb = b2a[t * 16 + n];
#pragma unroll
    for (int j = 0; j < 8; ++j) c1[t][j] = bb;
#pragma unroll
    for (int s = 0; s < 5; ++s) {
      v16h b = ((const v16h*)(fW2a + (size_t)(t * 5 + s) * 512))[lane];
      c1[t] = WMMA_F16(az[s], b, c1[t]);
    }
  }

  // ReLU + restage h1 into LDS (overwrites z; az already in registers).
#pragma unroll
  for (int t = 0; t < 16; ++t) {
#pragma unroll
    for (int j = 0; j < 8; ++j) {
      int M = j + 8 * g;
      float v = c1[t][j];
      zh[wave][M][t * 16 + n] = (_Float16)(v > 0.0f ? v : 0.0f);
    }
  }

  // Preload 8 A fragments of h1.
  v16h a2[8];
#pragma unroll
  for (int s = 0; s < 8; ++s) {
    const _Float16* rp = &zh[wave][n][0];
    int K0 = s * 32 + 8 * g;
    v8h_t lo = *(const v8h_t*)(rp + K0);
    v8h_t hi = *(const v8h_t*)(rp + K0 + 16);
#pragma unroll
    for (int i = 0; i < 8; ++i) { a2[s][i] = lo[i]; a2[s][i + 8] = hi[i]; }
  }

  // GEMM2: 32 N-tiles x 8 K-steps -> f32 output (coalesced 64B runs per group).
#pragma unroll
  for (int t = 0; t < 32; ++t) {
    v8f d;
    float bb = b2b[t * 16 + n];
#pragma unroll
    for (int j = 0; j < 8; ++j) d[j] = bb;
#pragma unroll
    for (int s = 0; s < 8; ++s) {
      v16h b = ((const v16h*)(fW2b + (size_t)(t * 8 + s) * 512))[lane];
      d = WMMA_F16(a2[s], b, d);
    }
#pragma unroll
    for (int j = 0; j < 8; ++j) {
      int M = j + 8 * g;
      long nd = (long)tile * 16 + M;
      if (nd < N) out[nd * 512 + t * 16 + n] = d[j];
    }
  }
}

// ---------------------------------------------------------------------------
extern "C" void kernel_launch(void* const* d_in, const int* in_sizes, int n_in,
                              void* d_out, int out_size, void* d_ws, size_t ws_size,
                              hipStream_t stream) {
  const float* x     = (const float*)d_in[0];
  const int*   eidx  = (const int*)d_in[1];
  const float* eattr = (const float*)d_in[2];
  const float* u     = (const float*)d_in[3];
  const int*   batch = (const int*)d_in[4];
  const float* W1a = (const float*)d_in[5];  const float* b1a = (const float*)d_in[6];
  const float* W1b = (const float*)d_in[7];  const float* b1b = (const float*)d_in[8];
  const float* W2a = (const float*)d_in[9];  const float* b2a = (const float*)d_in[10];
  const float* W2b = (const float*)d_in[11]; const float* b2b = (const float*)d_in[12];

  const int N = in_sizes[0] / 2;   // x: [N,2]
  const int E = in_sizes[2] / 2;   // edge_attr: [E,2]
  float* out = (float*)d_out;

  // Workspace layout.
  char* ws = (char*)d_ws;
  size_t off = 0;
  float* sums = (float*)(ws + off); off += (size_t)N * 128 * sizeof(float);
  float* cnt  = (float*)(ws + off); off += (size_t)N * sizeof(float);
  off = (off + 255) & ~(size_t)255;
  _Float16* fW1a = (_Float16*)(ws + off); off += (size_t)4   * 512 * 2;  // 1 kstep x 4 ntiles
  _Float16* fW1b = (_Float16*)(ws + off); off += (size_t)16  * 512 * 2;  // 2 x 8
  _Float16* fW2a = (_Float16*)(ws + off); off += (size_t)80  * 512 * 2;  // 5 x 16
  _Float16* fW2b = (_Float16*)(ws + off); off += (size_t)256 * 512 * 2;  // 8 x 32

  // 1) Zero scatter accumulators (sums and cnt are contiguous).
  long zn = (long)N * 128 + N;
  zero_kernel<<<1024, 256, 0, stream>>>(sums, zn);

  // 2) Pack weights into WMMA fragment order (f16).
  pack_kernel<<<(4   * 512 + 255) / 256, 256, 0, stream>>>(W1a, fW1a, 4,   64,  1, 4   * 512);
  pack_kernel<<<(16  * 512 + 255) / 256, 256, 0, stream>>>(W1b, fW1b, 64,  128, 2, 16  * 512);
  pack_kernel<<<(80  * 512 + 255) / 256, 256, 0, stream>>>(W2a, fW2a, 132, 256, 5, 80  * 512);
  pack_kernel<<<(256 * 512 + 255) / 256, 256, 0, stream>>>(W2b, fW2b, 256, 512, 8, 256 * 512);

  // 3) Edge MLP + scatter-add (one 16-edge tile per wave, 4 waves/block).
  int etiles  = (E + 15) / 16;
  int eblocks = (etiles + 3) / 4;
  edge_kernel<<<eblocks, 128, 0, stream>>>(x, eidx, eattr, b1a, b1b,
                                           fW1a, fW1b, sums, cnt, E);

  // 4) Node MLP (one 16-node tile per wave, 4 waves/block).
  int ntiles  = (N + 15) / 16;
  int nblocks = (ntiles + 3) / 4;
  node_kernel<<<nblocks, 128, 0, stream>>>(x, u, batch, sums, cnt,
                                           b2a, b2b, fW2a, fW2b, out, N);
}
